// WMN_489626272220
// MI455X (gfx1250) — compile-verified, matrix operand order checked by reading, not run on hardware
//
#include <hip/hip_runtime.h>
#include <hip/hip_bf16.h>

typedef __attribute__((ext_vector_type(16))) __bf16 v16bf;
typedef __attribute__((ext_vector_type(8)))  float  v8f;
typedef __attribute__((ext_vector_type(4)))  unsigned int u32x4;
typedef __attribute__((ext_vector_type(8)))  int i32x8;
typedef __attribute__((ext_vector_type(4)))  int i32x4;

#define B_      32
#define MEM_    50
#define SEN_    20
#define QLEN_   20
#define H_      256
#define H3_     768
#define VOCAB_  100000
#define HOPS_   3

// ---------------- WMMA fragment helpers ----------------
// CDNA5 16-bit A/B layout: lanes 0-15 take K {0..7,16..23}, lanes 16-31 take
// K {8..15,24..31}; row (A) / col (B) = lane%16.
__device__ __forceinline__ v16bf frag_from_f32(float4 a0, float4 a1, float4 b0, float4 b1) {
  v16bf r;
  r[0]=(__bf16)a0.x;  r[1]=(__bf16)a0.y;  r[2]=(__bf16)a0.z;  r[3]=(__bf16)a0.w;
  r[4]=(__bf16)a1.x;  r[5]=(__bf16)a1.y;  r[6]=(__bf16)a1.z;  r[7]=(__bf16)a1.w;
  r[8]=(__bf16)b0.x;  r[9]=(__bf16)b0.y;  r[10]=(__bf16)b0.z; r[11]=(__bf16)b0.w;
  r[12]=(__bf16)b1.x; r[13]=(__bf16)b1.y; r[14]=(__bf16)b1.z; r[15]=(__bf16)b1.w;
  return r;
}

__device__ __forceinline__ v16bf frag_f32mem(const float* rowp, int kb0, int kb1) {
  float4 a0 = *(const float4*)(rowp + kb0);
  float4 a1 = *(const float4*)(rowp + kb0 + 4);
  float4 a2 = *(const float4*)(rowp + kb1);
  float4 a3 = *(const float4*)(rowp + kb1 + 4);
  return frag_from_f32(a0, a1, a2, a3);
}

// ---------------- Tensor Data Mover: 2-D f32 tile -> LDS ----------------
// D# per CDNA5 ISA ch8: group0 {count=1, lds_addr, global_addr, type=2},
// group1 {data_size=4B, pad every 32 DW by 4 DW, tensor=[32 x tensor_rows],
// tile=[32 x tile_rows], dim0_stride=row_stride}. Rows past tensor_rows read
// as zero (free edge padding). LDS row pitch becomes 36 floats (144B).
#define SP 36

__device__ __forceinline__ unsigned lds_off(const void* p) {
  return (unsigned)(unsigned long long)p;   // low 32 bits of flat = LDS byte addr
}

__device__ __forceinline__ void tdm_load_2d(const void* gsrc, unsigned lds_byte,
                                            int tile_rows, int tensor_rows,
                                            int row_stride_elems) {
  const unsigned long long ga = (unsigned long long)gsrc;
  u32x4 g0;
  g0[0] = 1u;                                              // count=1, user desc
  g0[1] = lds_byte;                                        // LDS dest (bytes)
  g0[2] = (unsigned)ga;                                    // global addr lo
  g0[3] = (unsigned)((ga >> 32) & 0x1ffffffull) | (2u << 30); // addr hi | type=2
  i32x8 g1;
  g1[0] = (2 << 16) | (1 << 20) | (4 << 22) | (3 << 25);   // 4B, pad 32DW->+4DW
  g1[1] = (int)(32u << 16);                                // tensor_dim0 = 32
  g1[2] = (int)(((unsigned)tensor_rows & 0xffffu) << 16);  // tensor_dim1 lo
  g1[3] = (int)((((unsigned)tensor_rows >> 16) & 0xffffu) | (32u << 16)); // |tile_dim0
  g1[4] = tile_rows & 0xffff;                              // tile_dim1, tile_dim2=0
  g1[5] = row_stride_elems;                                // dim0_stride lo32
  g1[6] = 0;
  g1[7] = 0;
  const i32x4 z4 = {0, 0, 0, 0};
  const i32x8 z8 = {0, 0, 0, 0, 0, 0, 0, 0};
  __builtin_amdgcn_tensor_load_to_lds(g0, g1, z4, z4, z8, 0);
}

// ---------------- generic GEMM: C[M,N] = act(A[M,K] @ Bw[N,K]^T + bias) ----------------
// TDM double-buffered staging: wave 0 DMAs the next 64x32 (A) and 128x32 (B)
// f32 tiles into LDS while all 8 waves run WMMAs on the current tiles.
#define GBM 64
#define GBN 128
#define GBK 32

__global__ __launch_bounds__(256) void gemm_bf16(
    const float* __restrict__ A, const float* __restrict__ Bw,
    const float* __restrict__ bias, float* __restrict__ C,
    int M, int N, int K, int act)
{
  __shared__ float As[2][GBM * SP];
  __shared__ float Bs[2][GBN * SP];
  const int tid  = threadIdx.x;
  const int lane = tid & 31;
  const int wave = tid >> 5;      // 0..7
  const int m_t  = wave & 3;      // output m-tile of this wave
  const int n_g  = wave >> 2;     // n-tile group (4 tiles each)
  const int hsel = lane >> 4;     // 0/1: K-half pattern
  const int l16  = lane & 15;
  const int kb0  = hsel * 8;
  const int kb1  = kb0 + 16;
  const int block_m = blockIdx.y * GBM;
  const int block_n = blockIdx.x * GBN;
  const bool issuer = (tid < 32);

  v8f acc[4] = {};

  if (issuer) {  // prologue: DMA first k-block into buffer 0
    tdm_load_2d(A  + (size_t)block_m * K, lds_off(&As[0][0]), GBM, M - block_m, K);
    tdm_load_2d(Bw + (size_t)block_n * K, lds_off(&Bs[0][0]), GBN, N - block_n, K);
  }

  int cur = 0;
  for (int k0 = 0; k0 < K; k0 += GBK) {
    if (issuer) {
      if (k0 + GBK < K) {  // DMA next k-block into the other buffer
        tdm_load_2d(A  + (size_t)block_m * K + (k0 + GBK), lds_off(&As[cur ^ 1][0]),
                    GBM, M - block_m, K);
        tdm_load_2d(Bw + (size_t)block_n * K + (k0 + GBK), lds_off(&Bs[cur ^ 1][0]),
                    GBN, N - block_n, K);
        __builtin_amdgcn_s_wait_tensorcnt(2);   // current pair landed, next in flight
      } else {
        __builtin_amdgcn_s_wait_tensorcnt(0);
      }
    }
    __syncthreads();   // release all waves onto the landed tiles

    v16bf afrag = frag_f32mem(&As[cur][(m_t * 16 + l16) * SP], kb0, kb1);
#pragma unroll
    for (int q = 0; q < 4; ++q) {
      v16bf bfrag = frag_f32mem(&Bs[cur][((n_g * 4 + q) * 16 + l16) * SP], kb0, kb1);
      acc[q] = __builtin_amdgcn_wmma_f32_16x16x32_bf16(
          false, afrag, false, bfrag, (short)0, acc[q], false, false);
    }
    __syncthreads();   // all reads done before this buffer is DMA-overwritten
    cur ^= 1;
  }

  // C/D layout: VGPR j -> (M = j + 8*hsel, N = l16) within 16x16 tile
#pragma unroll
  for (int q = 0; q < 4; ++q) {
    const int col = block_n + (n_g * 4 + q) * 16 + l16;
    if (col < N) {
      const float bv = bias ? bias[col] : 0.f;
#pragma unroll
      for (int j = 0; j < 8; ++j) {
        const int row = block_m + m_t * 16 + hsel * 8 + j;
        if (row < M) {
          float v = acc[q][j] + bv;
          if (act) v = fmaxf(v, 0.f);
          C[(size_t)row * N + col] = v;
        }
      }
    }
  }
}

// ---------------- persistent-workgroup GRU scan ----------------
// One workgroup, 32 waves. h [64 x H] f32 in LDS (pitch 260 -> 16B aligned,
// bank-conflict-free fragment loads). Wave (m_t, g): m-tile m_t, hidden cols
// [32g, 32g+32). Owns accumulator tiles {2g+tt + gate*16} so the (r,z,n)
// triple for each hidden column lives in one wave's registers.
#define HP 260

__global__ __launch_bounds__(1024) void gru_scan(
    const float* __restrict__ gi,    // precomputed x@w_ih.T+b_ih rows [*,768]
    const float* __restrict__ w_hh,  // [768,256] f32
    const float* __restrict__ b_hh,  // [768]
    const int*   __restrict__ sel,   // [outer*rows] pick-timestep per row
    float* __restrict__ outp,        // [outer*rows, 256]
    int rows, int T, int outer, int gi_outer_stride, int gi_row_stride)
{
  extern __shared__ float hsm[];     // 64 * HP floats
  const int tid  = threadIdx.x;
  const int lane = tid & 31;
  const int wave = tid >> 5;         // 0..31
  const int m_t  = wave & 3;
  const int g    = wave >> 2;        // 0..7
  const int hsel = lane >> 4;
  const int l16  = lane & 15;
  const int kb0  = hsel * 8;
  const int kb1  = kb0 + 16;

  for (int i = tid; i < 64 * HP; i += 1024) hsm[i] = 0.f;
  __syncthreads();

  for (int b = 0; b < outer; ++b) {
    for (int t = 0; t < T; ++t) {
      v8f acc[6] = {};
#pragma unroll
      for (int k0 = 0; k0 < H_; k0 += 32) {
        v16bf afrag = frag_f32mem(&hsm[(m_t * 16 + l16) * HP + k0], kb0, kb1);
#pragma unroll
        for (int q = 0; q < 6; ++q) {
          const int gate = q >> 1, tt = q & 1;
          const int nt = 2 * g + tt + gate * 16;     // n-tile in [0,48)
          v16bf bfrag = frag_f32mem(w_hh + (size_t)(nt * 16 + l16) * H_ + k0, kb0, kb1);
          acc[q] = __builtin_amdgcn_wmma_f32_16x16x32_bf16(
              false, afrag, false, bfrag, (short)0, acc[q], false, false);
        }
      }
      __syncthreads();   // all reads of h done before updates
#pragma unroll
      for (int tt = 0; tt < 2; ++tt) {
        const int c = 32 * g + 16 * tt + l16;        // hidden column 0..255
        const float bhr = b_hh[c];
        const float bhz = b_hh[H_ + c];
        const float bhn = b_hh[2 * H_ + c];
#pragma unroll
        for (int j = 0; j < 8; ++j) {
          const int m = m_t * 16 + hsel * 8 + j;
          float i_r = 0.f, i_z = 0.f, i_n = 0.f;
          const bool valid = (m < rows);
          if (valid) {
            const float* gp = gi +
              ((size_t)b * gi_outer_stride + (size_t)m * gi_row_stride + t) * H3_;
            i_r = gp[c]; i_z = gp[H_ + c]; i_n = gp[2 * H_ + c];
          }
          const float h_r = acc[0 + tt][j] + bhr;
          const float h_z = acc[2 + tt][j] + bhz;
          const float h_n = acc[4 + tt][j] + bhn;
          const float r = 1.f / (1.f + __expf(-(i_r + h_r)));
          const float z = 1.f / (1.f + __expf(-(i_z + h_z)));
          const float n = tanhf(i_n + r * h_n);
          const float hold = hsm[m * HP + c];
          const float hnew = (1.f - z) * n + z * hold;
          hsm[m * HP + c] = hnew;                    // (m,c) owned by this wave
          if (valid && sel[b * rows + m] == t)
            outp[((size_t)(b * rows + m)) * H_ + c] = hnew;
        }
      }
      __syncthreads();
    }
  }
}

// ---------------- small kernels ----------------
__global__ void sidx_kernel(const int* __restrict__ story, const int* __restrict__ query,
                            int* __restrict__ sel) {
  const int i = blockIdx.x * blockDim.x + threadIdx.x;
  if (i >= B_ * MEM_ + B_) return;
  const int* tok = (i < B_ * MEM_) ? (story + (size_t)i * SEN_)
                                   : (query + (size_t)(i - B_ * MEM_) * QLEN_);
  int idx = SEN_ - 1;
  for (int t = SEN_ - 1; t >= 0; --t) if (tok[t] == 2) idx = t;
  sel[i] = idx;
}

__global__ void gather_kernel(const int* __restrict__ story, const int* __restrict__ query,
                              const float* __restrict__ embed, float* __restrict__ x) {
  const int gidx = blockIdx.x * blockDim.x + threadIdx.x;
  const int row = gidx >> 6;
  const int c4  = (gidx & 63) * 4;
  const int NTOK = B_ * MEM_ * SEN_ + B_ * QLEN_;
  if (row >= NTOK) return;
  const int tok = (row < B_ * MEM_ * SEN_) ? story[row] : query[row - B_ * MEM_ * SEN_];
  float4 v = *(const float4*)(embed + (size_t)tok * H_ + c4);
  *(float4*)(x + (size_t)row * H_ + c4) = v;
}

__global__ __launch_bounds__(128) void attn_kernel(
    const float* __restrict__ q, const float* __restrict__ kmem,
    const float* __restrict__ vmem, float* __restrict__ outp) {
  __shared__ float prob[4][64];
  const int b = blockIdx.x;
  const int h = threadIdx.x >> 5;
  const int lane = threadIdx.x & 31;
  const float* qh = q + b * H_ + h * 64;
  float sc[2];
#pragma unroll
  for (int u = 0; u < 2; ++u) {
    const int m = lane + 32 * u;
    if (m < MEM_) {
      const float* kp = kmem + ((size_t)b * MEM_ + m) * H_ + h * 64;
      float a = 0.f;
      for (int d = 0; d < 64; ++d) a += kp[d] * qh[d];
      sc[u] = a * 0.125f;                         // 1/sqrt(dk=64)
    } else sc[u] = -1e30f;
  }
  float mx = fmaxf(sc[0], sc[1]);
  for (int off = 16; off; off >>= 1) mx = fmaxf(mx, __shfl_xor(mx, off, 32));
  const float e0 = __expf(sc[0] - mx);
  const float e1 = (lane + 32 < MEM_) ? __expf(sc[1] - mx) : 0.f;
  float s = e0 + e1;
  for (int off = 16; off; off >>= 1) s += __shfl_xor(s, off, 32);
  const float inv = 1.f / s;
  prob[h][lane] = e0 * inv;
  prob[h][lane + 32] = e1 * inv;
  __syncthreads();
#pragma unroll
  for (int u = 0; u < 2; ++u) {
    const int d = lane + 32 * u;
    float a = 0.f;
    for (int m = 0; m < MEM_; ++m)
      a += prob[h][m] * vmem[((size_t)b * MEM_ + m) * H_ + h * 64 + d];
    outp[b * H_ + h * 64 + d] = a;
  }
}

__global__ void combine_kernel(const float* __restrict__ mb /*[3][32][256]*/,
                               const float* __restrict__ rq, float* __restrict__ comb) {
  const int gidx = blockIdx.x * blockDim.x + threadIdx.x;
  if (gidx >= B_ * 9 * H3_) return;
  const int rc = gidx / H3_, c = gidx % H3_;
  const int b = rc / 9, p = rc % 9;
  const int i = p / 3, j = p % 3;
  float v;
  if (c < H_)            v = mb[((size_t)j * B_ + b) * H_ + c];          // left = hop j
  else if (c < 2 * H_)   v = mb[((size_t)i * B_ + b) * H_ + (c - H_)];   // right = hop i
  else                   v = rq[b * H_ + (c - 2 * H_)];
  comb[gidx] = v;
}

__global__ void rsum_kernel(const float* __restrict__ t2, float* __restrict__ outp) {
  const int gidx = blockIdx.x * blockDim.x + threadIdx.x;
  if (gidx >= B_ * H3_) return;
  const int b = gidx / H3_, c = gidx % H3_;
  float a = 0.f;
  for (int p = 0; p < 9; ++p) a += fmaxf(t2[((size_t)(b * 9 + p)) * H3_ + c], 0.f);
  outp[gidx] = a;
}

// ---------------- host orchestration ----------------
extern "C" void kernel_launch(void* const* d_in, const int* in_sizes, int n_in,
                              void* d_out, int out_size, void* d_ws, size_t ws_size,
                              hipStream_t stream) {
  (void)in_sizes; (void)n_in; (void)out_size; (void)ws_size;
  const int*   story = (const int*)d_in[0];
  const int*   query = (const int*)d_in[1];
  const float* embed = (const float*)d_in[2];
  const float* w_ih  = (const float*)d_in[3];
  const float* w_hh  = (const float*)d_in[4];
  const float* b_ih  = (const float*)d_in[5];
  const float* b_hh  = (const float*)d_in[6];
  const float* ft_w1 = (const float*)d_in[7];
  const float* ft_b1 = (const float*)d_in[8];
  const float* ft_w2 = (const float*)d_in[9];
  const float* ft_b2 = (const float*)d_in[10];
  const float* wq = (const float*)d_in[11];
  const float* bq = (const float*)d_in[12];
  const float* wk = (const float*)d_in[13];
  const float* bk = (const float*)d_in[14];
  const float* wv = (const float*)d_in[15];
  const float* bv = (const float*)d_in[16];
  const float* wo = (const float*)d_in[17];
  const float* bo = (const float*)d_in[18];
  const float* g_w1 = (const float*)d_in[19];
  const float* g_b1 = (const float*)d_in[20];
  const float* g_w2 = (const float*)d_in[21];
  const float* g_b2 = (const float*)d_in[22];
  const float* v_w  = (const float*)d_in[23];
  float* logits = (float*)d_out;

  char* ws = (char*)d_ws;
  size_t cur = 0;
  auto alloc = [&](size_t bytes) -> void* {
    void* p = ws + cur;
    cur = (cur + bytes + 255) & ~(size_t)255;
    return p;
  };
  const int NTOK = B_ * MEM_ * SEN_ + B_ * QLEN_;           // 32640
  float* x_emb   = (float*)alloc((size_t)NTOK * H_ * 4);
  float* gi      = (float*)alloc((size_t)NTOK * H3_ * 4);
  int*   sel     = (int*)  alloc((size_t)(B_ * MEM_ + B_) * 4);
  float* memslot = (float*)alloc((size_t)B_ * MEM_ * H_ * 4);
  float* realq   = (float*)alloc((size_t)B_ * H_ * 4);
  float* kmem    = (float*)alloc((size_t)B_ * MEM_ * H_ * 4);
  float* vmem    = (float*)alloc((size_t)B_ * MEM_ * H_ * 4);
  float* ft_tmp  = (float*)alloc((size_t)B_ * H_ * 4);
  float* okbuf   = (float*)alloc((size_t)B_ * H_ * 4);
  float* qproj   = (float*)alloc((size_t)B_ * H_ * 4);
  float* attout  = (float*)alloc((size_t)B_ * H_ * 4);
  float* mbbuf   = (float*)alloc((size_t)HOPS_ * B_ * H_ * 4);
  float* comb    = (float*)alloc((size_t)B_ * 9 * H3_ * 4);
  float* t1      = (float*)alloc((size_t)B_ * 9 * H3_ * 4);
  float* t2      = (float*)alloc((size_t)B_ * 9 * H3_ * 4);
  float* reason  = (float*)alloc((size_t)B_ * H3_ * 4);

  auto gemm = [&](const float* A, const float* Bw, const float* bias, float* C,
                  int M, int N, int K, int act) {
    dim3 grid((N + GBN - 1) / GBN, (M + GBM - 1) / GBM);
    gemm_bf16<<<grid, 256, 0, stream>>>(A, Bw, bias, C, M, N, K, act);
  };

  // 1) SEQEND indices for all story sentences + queries
  sidx_kernel<<<(B_ * MEM_ + B_ + 255) / 256, 256, 0, stream>>>(story, query, sel);
  // 2) embedding gather for all tokens (story then query)
  gather_kernel<<<(NTOK * 64 + 255) / 256, 256, 0, stream>>>(story, query, embed, x_emb);
  // 3) hoisted input-gate GEMM: gi = x @ w_ih.T + b_ih  (M=32640, the big parallel GEMM)
  gemm(x_emb, w_ih, b_ih, gi, NTOK, H3_, H_, 0);
  // 4) sequential GRU scans (persistent single workgroup, h in LDS)
  const size_t smem = 64 * HP * sizeof(float);
  gru_scan<<<1, 1024, smem, stream>>>(gi, w_hh, b_hh, sel, memslot,
                                      MEM_, SEN_, B_, MEM_ * SEN_, SEN_);
  gru_scan<<<1, 1024, smem, stream>>>(gi + (size_t)B_ * MEM_ * SEN_ * H3_, w_hh, b_hh,
                                      sel + B_ * MEM_, realq, B_, QLEN_, 1, 0, QLEN_);
  // 5) K/V projections of memory (hop-invariant, computed once)
  gemm(memslot, wk, bk, kmem, B_ * MEM_, H_, H_, 0);
  gemm(memslot, wv, bv, vmem, B_ * MEM_, H_, H_, 0);
  // 6) o_k = f_t(real_query)
  gemm(realq, ft_w1, ft_b1, ft_tmp, B_, H_, H_, 1);
  gemm(ft_tmp, ft_w2, ft_b2, okbuf, B_, H_, H_, 0);
  // 7) multi-hop attention
  for (int hop = 0; hop < HOPS_; ++hop) {
    gemm(okbuf, wq, bq, qproj, B_, H_, H_, 0);
    attn_kernel<<<B_, 128, 0, stream>>>(qproj, kmem, vmem, attout);
    float* mb = mbbuf + (size_t)hop * B_ * H_;
    gemm(attout, wo, bo, mb, B_, H_, H_, 0);
    if (hop + 1 < HOPS_) {
      gemm(mb, ft_w1, ft_b1, ft_tmp, B_, H_, H_, 1);
      gemm(ft_tmp, ft_w2, ft_b2, okbuf, B_, H_, H_, 0);
    }
  }
  // 8) reason module
  combine_kernel<<<(B_ * 9 * H3_ + 255) / 256, 256, 0, stream>>>(mbbuf, realq, comb);
  gemm(comb, g_w1, g_b1, t1, B_ * 9, H3_, H3_, 1);
  gemm(t1, g_w2, g_b2, t2, B_ * 9, H3_, H3_, 0);
  rsum_kernel<<<(B_ * H3_ + 255) / 256, 256, 0, stream>>>(t2, reason);
  // 9) vocab projection (bandwidth-bound on v_w: f32 read, in-register bf16 cvt)
  gemm(reason, v_w, nullptr, logits, B_, VOCAB_, H3_, 0);
}
